// Model_52888227283443
// MI455X (gfx1250) — matrix operand order repaired and checked by reading, not compiled, not run on hardware
//
#include <hip/hip_runtime.h>
#include <hip/hip_bf16.h>

typedef __bf16 bf16_t;
typedef __attribute__((ext_vector_type(16))) __bf16 v16bf;
typedef __attribute__((ext_vector_type(8)))  __bf16 v8bf;
typedef __attribute__((ext_vector_type(8)))  float  v8f;
typedef __attribute__((ext_vector_type(4)))  int    v4i;

#if defined(__has_builtin)
#if __has_builtin(__builtin_amdgcn_global_load_async_to_lds_b128)
#define HAVE_ASYNC_LDS 1
#endif
#endif
#ifndef HAVE_ASYNC_LDS
#define HAVE_ASYNC_LDS 0
#endif

#if HAVE_ASYNC_LDS
typedef __attribute__((address_space(1))) v4i* as1_v4ip;
typedef __attribute__((address_space(3))) v4i* as3_v4ip;
// Async DMA-style copy of 16 bytes per active lane: global -> LDS (ASYNCcnt).
static __device__ __forceinline__ void async_copy_b128(const void* g, void* l) {
  __builtin_amdgcn_global_load_async_to_lds_b128((as1_v4ip)g, (as3_v4ip)l, 0, 0);
}
#endif

static __device__ __forceinline__ void async_wait0() {
#if defined(__has_builtin)
#if __has_builtin(__builtin_amdgcn_s_wait_asynccnt)
  __builtin_amdgcn_s_wait_asynccnt(0);
#else
  asm volatile("s_wait_asynccnt 0" ::: "memory");
#endif
#else
  asm volatile("s_wait_asynccnt 0" ::: "memory");
#endif
}

static __device__ __forceinline__ v8f zerov8() {
  v8f z = {0.f, 0.f, 0.f, 0.f, 0.f, 0.f, 0.f, 0.f};
  return z;
}

static __device__ __forceinline__ v8f wmma_bf16(v16bf a, v16bf b, v8f c) {
  return __builtin_amdgcn_wmma_f32_16x16x32_bf16(false, a, false, b, (short)0, c,
                                                 false, false);
}

// A fragment (16x32 bf16), source row-major with given stride (halfs).
static __device__ __forceinline__ v16bf load_a_frag(const bf16_t* base, int stride,
                                                    int k0, int lane) {
  const int m = lane & 15;
  const int khalf = (lane >> 4) << 3;
  const bf16_t* p = base + m * stride + k0 + khalf;
  v8bf lo = *(const v8bf*)(p);
  v8bf hi = *(const v8bf*)(p + 16);
  v16bf r;
#pragma unroll
  for (int i = 0; i < 8; ++i) { r[i] = lo[i]; r[8 + i] = hi[i]; }
  return r;
}

// B fragment (32x16 bf16) where B[k][n] = base[n*stride + k0 + k] (contiguous in k).
static __device__ __forceinline__ v16bf load_b_kcontig(const bf16_t* base, int stride,
                                                       int k0, int lane) {
  const int n = lane & 15;
  const int kb = k0 + ((lane >> 4) << 3);
  const bf16_t* p = base + n * stride + kb;
  v8bf a = *(const v8bf*)p;
  v8bf b = *(const v8bf*)(p + 16);
  v16bf r;
#pragma unroll
  for (int i = 0; i < 8; ++i) { r[i] = a[i]; r[8 + i] = b[i]; }
  return r;
}

// B fragment where only k=0..15 is valid (attention d_k=16 padded to K=32).
static __device__ __forceinline__ v16bf load_b_rows16(const bf16_t* base, int stride,
                                                      int col0, int lane) {
  const int n = lane & 15;
  v16bf r;
  const bf16_t* p = base + n * stride + col0 + ((lane >> 4) << 3);
  v8bf a = *(const v8bf*)p;
#pragma unroll
  for (int i = 0; i < 8; ++i) { r[i] = a[i]; r[8 + i] = (bf16_t)0.0f; }
  return r;
}

// Pre-packed weights: tile-block layout, block = (kt,nt), 32 lanes x 16 halfs.
static __device__ __forceinline__ v16bf load_b_packed(const bf16_t* pw, int kt, int nt,
                                                      int ntiles, int lane) {
  const bf16_t* p = pw + (((size_t)(kt * ntiles + nt)) * 32 + lane) * 16;
  v8bf a = *(const v8bf*)p;
  v8bf b = *(const v8bf*)(p + 8);
  v16bf r;
#pragma unroll
  for (int i = 0; i < 8; ++i) { r[i] = a[i]; r[8 + i] = b[i]; }
  return r;
}

// ---------------------------------------------------------------------------
// Pack f32 weight [Ksrc x Nsrc] into padded [Kpad x Npad] bf16 B-fragment blocks.
// ---------------------------------------------------------------------------
__global__ __launch_bounds__(256) void pack_kernel(const float* __restrict__ src,
                                                   bf16_t* __restrict__ dst,
                                                   int Ksrc, int Nsrc, int Kpad, int Npad) {
  const int idx = blockIdx.x * 256 + threadIdx.x;
  if (idx >= Kpad * Npad) return;
  const int k = idx / Npad;
  const int n = idx - k * Npad;
  const float v = (k < Ksrc && n < Nsrc) ? src[k * Nsrc + n] : 0.0f;
  const int kt = k >> 5, kin = k & 31;
  const int nt = n >> 4, nin = n & 15;
  const int lane = nin + (((kin >> 3) & 1) ? 16 : 0);
  const int e = (kin & 7) + ((kin >> 4) << 3);
  const size_t o = (((size_t)(kt * (Npad >> 4) + nt)) * 32 + lane) * 16 + e;
  dst[o] = (bf16_t)v;
}

// ---------------------------------------------------------------------------
// Convert embedding table [rows x 300] f32 -> bf16 padded to stride 320
// (30 MB: L2-resident thereafter; enables zero-VALU async gather to LDS).
// ---------------------------------------------------------------------------
__global__ __launch_bounds__(256) void cvt_emb_kernel(const float* __restrict__ src,
                                                      bf16_t* __restrict__ dst,
                                                      int rows) {
  const size_t idx = (size_t)blockIdx.x * 256 + threadIdx.x;
  if (idx >= (size_t)rows * 320) return;
  const size_t r = idx / 320;
  const int c = (int)(idx - r * 320);
  dst[idx] = (c < 300) ? (bf16_t)src[r * 300 + c] : (bf16_t)0.0f;
}

// ---------------------------------------------------------------------------
// Fused encoder: X -> MHSA (16 heads x d_k=16) -> additive-attention pool.
// One workgroup (8 wave32) per sequence.
// ---------------------------------------------------------------------------
template <int MT, int KX, int SEQ, bool GATHER>
__global__ __launch_bounds__(256) void encoder_kernel(
    const int* __restrict__ tokA, int nA, const int* __restrict__ tokB,
    const bf16_t* __restrict__ embb,     // [V][KX] bf16, padded (GATHER only)
    const float* __restrict__ rows_in,   // [nseq][SEQ][256]   (!GATHER only)
    const bf16_t* __restrict__ pWq, const bf16_t* __restrict__ pWk,
    const bf16_t* __restrict__ pWv,
    const float* __restrict__ bq, const float* __restrict__ bk,
    const float* __restrict__ bv,
    const bf16_t* __restrict__ pA1,      // packed 256 x 208
    const float* __restrict__ b1,        // [200]
    const float* __restrict__ A2,        // [200]
    float* __restrict__ outA, float* __restrict__ outB) {
  constexpr int M = MT * 16;
  constexpr int KS = KX / 32;   // k-steps for QKV projections
  constexpr int D = 256;
  constexpr int NT = D / 16;    // 16 column tiles
  constexpr int NT1 = 13;       // pool hidden: 200 padded to 208 -> 13 tiles

  __shared__ bf16_t sXC[M * KX];       // X (stride KX), later ctx (stride 256)
  __shared__ bf16_t sQ[M * D];         // Q row-major
  __shared__ bf16_t sK[M * D];         // K row-major
  __shared__ bf16_t sVt[D * M];        // V transposed: [d][token]
  __shared__ bf16_t sAttn[8 * M * M];  // per-wave attn scratch
  __shared__ float  sA[M];
  __shared__ float  sAlpha[M];

  const int tid  = threadIdx.x;
  const int lane = tid & 31;
  const int wave = tid >> 5;
  const int seq  = blockIdx.x;
  const int r0   = (lane >> 4) << 3;  // D-fragment row offset (0 or 8)
  const int nlo  = lane & 15;         // D-fragment column within tile

  // ---- Stage 0: stage X into LDS as zero-padded bf16 ----
  for (int i = tid; i < M * KX; i += 256) sXC[i] = (bf16_t)0.0f;
  __syncthreads();
  if (GATHER) {
    const int* tok = (seq < nA) ? (tokA + (size_t)seq * 32)
                                : (tokB + (size_t)(seq - nA) * 32);
    // Each wave DMA-copies rows: KX*2 bytes = (KX/8) 16B chunks per row.
    for (int t = wave; t < SEQ; t += 8) {
      const char* g = (const char*)(embb + (size_t)tok[t] * KX);
      char* l = (char*)&sXC[t * KX];
#if HAVE_ASYNC_LDS
      async_copy_b128(g + lane * 16, l + lane * 16);
      constexpr int REM = (KX * 2 - 512) / 16;  // 16B chunks past the first 512B
      if (lane < REM) async_copy_b128(g + 512 + lane * 16, l + 512 + lane * 16);
#else
      for (int c = lane * 8; c < KX; c += 256)
        *(v8bf*)(l + c * 2) = *(const v8bf*)(g + c * 2);
#endif
    }
#if HAVE_ASYNC_LDS
    async_wait0();
#endif
  } else {
    const float* rows = rows_in + (size_t)seq * SEQ * D;
    for (int i = tid; i < SEQ * D; i += 256) {
      const int t = i >> 8, c = i & 255;
      sXC[t * KX + c] = (bf16_t)rows[i];
    }
  }
  __syncthreads();

  // ---- Stage 1: Q/K/V = X @ W + b (WMMA over packed weights) ----
  for (int t = wave; t < MT * NT; t += 8) {
    const int mt = t >> 4;
    const int nt = t & 15;
    v8f aq = zerov8(), ak = zerov8(), av = zerov8();
#pragma unroll 1
    for (int ks = 0; ks < KS; ++ks) {
      v16bf a = load_a_frag(sXC + mt * 16 * KX, KX, ks * 32, lane);
      aq = wmma_bf16(a, load_b_packed(pWq, ks, nt, NT, lane), aq);
      ak = wmma_bf16(a, load_b_packed(pWk, ks, nt, NT, lane), ak);
      av = wmma_bf16(a, load_b_packed(pWv, ks, nt, NT, lane), av);
    }
    const int n = nt * 16 + nlo;
    const float biasq = bq[n], biask = bk[n], biasv = bv[n];
#pragma unroll
    for (int r = 0; r < 8; ++r) {
      const int row = mt * 16 + r0 + r;
      sQ[row * D + n]  = (bf16_t)(aq[r] + biasq);
      sK[row * D + n]  = (bf16_t)(ak[r] + biask);
      sVt[n * M + row] = (bf16_t)(av[r] + biasv);
    }
  }
  __syncthreads();

  // ---- Stage 2: attention, 2 heads per wave ----
  bf16_t* sAw = sAttn + wave * M * M;
#pragma unroll 1
  for (int hh = 0; hh < 2; ++hh) {
    const int h = wave * 2 + hh;
    v8f sc[MT][MT];
#pragma unroll
    for (int i = 0; i < MT; ++i)
#pragma unroll
      for (int j = 0; j < MT; ++j) sc[i][j] = zerov8();

    // Q fragments: K dim = d_k (16 valid, padded to 32)
    v16bf aq[MT];
#pragma unroll
    for (int mt = 0; mt < MT; ++mt) {
      const int m = lane & 15;
      const int khalf = (lane >> 4) << 3;
      const bf16_t* p = sQ + (mt * 16 + m) * D + h * 16 + khalf;
      v8bf lo = *(const v8bf*)p;
      v16bf a;
#pragma unroll
      for (int i = 0; i < 8; ++i) { a[i] = lo[i]; a[8 + i] = (bf16_t)0.0f; }
      aq[mt] = a;
    }
#pragma unroll 1
    for (int nt = 0; nt < MT; ++nt) {
      v16bf b = load_b_rows16(sK + nt * 16 * D, D, h * 16, lane);
#pragma unroll
      for (int mt = 0; mt < MT; ++mt) sc[mt][nt] = wmma_bf16(aq[mt], b, sc[mt][nt]);
    }

    // softmax rows (reference: raw exp, denom + 1e-8); mask cols >= SEQ
#pragma unroll 1
    for (int mt = 0; mt < MT; ++mt) {
#pragma unroll
      for (int r = 0; r < 8; ++r) {
        float ev[MT];
        float esum = 0.0f;
#pragma unroll
        for (int nt = 0; nt < MT; ++nt) {
          const int c = nt * 16 + nlo;
          const float e = (c < SEQ) ? __expf(sc[mt][nt][r] * 0.25f) : 0.0f;
          ev[nt] = e;
          esum += e;
        }
#pragma unroll
        for (int o = 1; o < 16; o <<= 1) esum += __shfl_xor(esum, o, 16);
        const float inv = 1.0f / (esum + 1e-8f);
        const int row = mt * 16 + r0 + r;
#pragma unroll
        for (int nt = 0; nt < MT; ++nt)
          sAw[row * M + nt * 16 + nlo] = (bf16_t)(ev[nt] * inv);
      }
    }

    // ctx = attn @ V_h   (write into sXC with stride 256)
#pragma unroll 1
    for (int mt = 0; mt < MT; ++mt) {
      v8f c = zerov8();
#pragma unroll
      for (int ks = 0; ks < M / 32; ++ks) {
        v16bf a = load_a_frag(sAw + mt * 16 * M, M, ks * 32, lane);
        v16bf b = load_b_kcontig(sVt + (h * 16) * M, M, ks * 32, lane);
        c = wmma_bf16(a, b, c);
      }
      const int n = h * 16 + nlo;
#pragma unroll
      for (int r = 0; r < 8; ++r) {
        const int row = mt * 16 + r0 + r;
        sXC[row * D + n] = (bf16_t)c[r];
      }
    }
  }
  __syncthreads();

  // ---- Stage 3: additive-attention pool ----
  for (int i = tid; i < M; i += 256) sA[i] = 0.0f;
  __syncthreads();
#pragma unroll 1
  for (int t = wave; t < MT * NT1; t += 8) {
    const int mt = t / NT1;
    const int nt = t - mt * NT1;
    v8f acc = zerov8();
#pragma unroll 1
    for (int ks = 0; ks < 8; ++ks) {
      v16bf a = load_a_frag(sXC + mt * 16 * D, D, ks * 32, lane);
      v16bf b = load_b_packed(pA1, ks, nt, NT1, lane);
      acc = wmma_bf16(a, b, acc);
    }
    const int n = nt * 16 + nlo;
    const float b1v = (n < 200) ? b1[n] : 0.0f;
    const float a2v = (n < 200) ? A2[n] : 0.0f;
#pragma unroll
    for (int r = 0; r < 8; ++r) {
      const int row = mt * 16 + r0 + r;
      atomicAdd(&sA[row], tanhf(acc[r] + b1v) * a2v);
    }
  }
  __syncthreads();

  if (tid < 32) {
    const bool v0 = (tid < SEQ);
    const bool v1 = (32 + tid < SEQ) && (32 + tid < M);
    float a0 = v0 ? sA[tid] : -1e30f;
    float a1 = v1 ? sA[32 + tid] : -1e30f;
    float mx = fmaxf(a0, a1);
#pragma unroll
    for (int o = 16; o > 0; o >>= 1) mx = fmaxf(mx, __shfl_xor(mx, o, 32));
    const float e0 = v0 ? __expf(a0 - mx) : 0.0f;
    const float e1 = v1 ? __expf(a1 - mx) : 0.0f;
    float s = e0 + e1;
#pragma unroll
    for (int o = 16; o > 0; o >>= 1) s += __shfl_xor(s, o, 32);
    const float inv = 1.0f / s;
    sAlpha[tid] = e0 * inv;
    if (32 + tid < M) sAlpha[32 + tid] = e1 * inv;
  }
  __syncthreads();

  {
    const int d = tid;  // 256 threads, D == 256
    float acc = 0.0f;
#pragma unroll 1
    for (int r = 0; r < SEQ; ++r) acc += sAlpha[r] * (float)sXC[r * D + d];
    float* outp = GATHER ? ((seq < nA) ? (outA + (size_t)seq * D)
                                       : (outB + (size_t)(seq - nA) * D))
                         : (outA + (size_t)seq * D);
    outp[d] = acc;
  }
}

// ---------------------------------------------------------------------------
// y[b][n] = dot(r[b*5+n], u[b])
// ---------------------------------------------------------------------------
__global__ __launch_bounds__(256) void score_kernel(const float* __restrict__ r,
                                                    const float* __restrict__ u,
                                                    float* __restrict__ y) {
  const int b = blockIdx.x;
  const int lane = threadIdx.x & 31;
  const int w = threadIdx.x >> 5;
  if (w < 5) {
    const float* rv = r + (size_t)(b * 5 + w) * 256;
    const float* uv = u + (size_t)b * 256;
    float s = 0.0f;
    for (int i = lane; i < 256; i += 32) s += rv[i] * uv[i];
#pragma unroll
    for (int o = 16; o > 0; o >>= 1) s += __shfl_xor(s, o, 32);
    if (lane == 0) y[b * 5 + w] = s;
  }
}

// ---------------------------------------------------------------------------
extern "C" void kernel_launch(void* const* d_in, const int* in_sizes, int n_in,
                              void* d_out, int out_size, void* d_ws, size_t ws_size,
                              hipStream_t stream) {
  (void)in_sizes; (void)n_in; (void)out_size; (void)ws_size;
  const int*   hist = (const int*)d_in[0];
  const int*   samp = (const int*)d_in[1];
  const float* emb  = (const float*)d_in[2];
  const float* nWq  = (const float*)d_in[3];
  const float* nbq  = (const float*)d_in[4];
  const float* nWk  = (const float*)d_in[5];
  const float* nbk  = (const float*)d_in[6];
  const float* nWv  = (const float*)d_in[7];
  const float* nbv  = (const float*)d_in[8];
  const float* nA1  = (const float*)d_in[9];
  const float* nA1b = (const float*)d_in[10];
  const float* nA2  = (const float*)d_in[11];
  const float* uWq  = (const float*)d_in[13];
  const float* ubq  = (const float*)d_in[14];
  const float* uWk  = (const float*)d_in[15];
  const float* ubk  = (const float*)d_in[16];
  const float* uWv  = (const float*)d_in[17];
  const float* ubv  = (const float*)d_in[18];
  const float* uA1  = (const float*)d_in[19];
  const float* uA1b = (const float*)d_in[20];
  const float* uA2  = (const float*)d_in[21];
  float* y = (float*)d_out;

  char* ws = (char*)d_ws;
  size_t off = 0;
  auto alloc = [&](size_t bytes) -> void* {
    void* p = ws + off;
    off = (off + bytes + 255) & ~(size_t)255;
    return p;
  };
  bf16_t* pnWq = (bf16_t*)alloc(320 * 256 * sizeof(bf16_t));
  bf16_t* pnWk = (bf16_t*)alloc(320 * 256 * sizeof(bf16_t));
  bf16_t* pnWv = (bf16_t*)alloc(320 * 256 * sizeof(bf16_t));
  bf16_t* pnA1 = (bf16_t*)alloc(256 * 208 * sizeof(bf16_t));
  bf16_t* puWq = (bf16_t*)alloc(256 * 256 * sizeof(bf16_t));
  bf16_t* puWk = (bf16_t*)alloc(256 * 256 * sizeof(bf16_t));
  bf16_t* puWv = (bf16_t*)alloc(256 * 256 * sizeof(bf16_t));
  bf16_t* puA1 = (bf16_t*)alloc(256 * 208 * sizeof(bf16_t));
  bf16_t* embb = (bf16_t*)alloc((size_t)50000 * 320 * sizeof(bf16_t));
  float*  h    = (float*)alloc((size_t)6400 * 256 * sizeof(float));
  float*  rvec = (float*)alloc((size_t)640 * 256 * sizeof(float));
  float*  uvec = (float*)alloc((size_t)128 * 256 * sizeof(float));

  auto pack = [&](const float* s, bf16_t* d, int Ks, int Ns, int Kp, int Np) {
    const int tot = Kp * Np;
    pack_kernel<<<(tot + 255) / 256, 256, 0, stream>>>(s, d, Ks, Ns, Kp, Np);
  };
  pack(nWq, pnWq, 300, 256, 320, 256);
  pack(nWk, pnWk, 300, 256, 320, 256);
  pack(nWv, pnWv, 300, 256, 320, 256);
  pack(nA1, pnA1, 256, 200, 256, 208);
  pack(uWq, puWq, 256, 256, 256, 256);
  pack(uWk, puWk, 256, 256, 256, 256);
  pack(uWv, puWv, 256, 256, 256, 256);
  pack(uA1, puA1, 256, 200, 256, 208);

  {
    const size_t tot = (size_t)50000 * 320;
    cvt_emb_kernel<<<(int)((tot + 255) / 256), 256, 0, stream>>>(emb, embb, 50000);
  }

  // News encoder: 6400 hist + 640 samp sequences, S=30, E=300 (padded 320).
  encoder_kernel<2, 320, 30, true><<<7040, 256, 0, stream>>>(
      hist, 6400, samp, embb, nullptr,
      pnWq, pnWk, pnWv, nbq, nbk, nbv, pnA1, nA1b, nA2, h, rvec);

  // User encoder: 128 users, S=50, D=256.
  encoder_kernel<4, 256, 50, false><<<128, 256, 0, stream>>>(
      nullptr, 0, nullptr, nullptr, h,
      puWq, puWk, puWv, ubq, ubk, ubv, puA1, uA1b, uA2, uvec, nullptr);

  score_kernel<<<128, 256, 0, stream>>>(rvec, uvec, y);
}
